// LSTMModel_13288628814043
// MI455X (gfx1250) — compile-verified, matrix operand order checked by reading, not run on hardware
//
#include <hip/hip_runtime.h>
#include <hip/hip_bf16.h>

// ---------------------------------------------------------------------------
// Model dims (from reference)
// ---------------------------------------------------------------------------
#define Bb   64
#define Tt   288
#define Hh   512
#define G4   2048           // 4 * H gate width
#define NHd  4
#define HDd  32
#define D1d  256
#define D2d  128
#define FOTH 64
#define ROWS (Bb * Tt)      // 18432

typedef __attribute__((ext_vector_type(16))) _Float16 v16h;
typedef __attribute__((ext_vector_type(8)))  float    v8f;

__device__ __forceinline__ v8f wmma16(v16h a, v16h b, v8f c) {
  return __builtin_amdgcn_wmma_f32_16x16x32_f16(false, a, false, b, (short)0, c,
                                                false, false);
}

__device__ __forceinline__ float sigf(float x) {
  return 1.f / (1.f + __expf(-x));
}

// ---------------------------------------------------------------------------
// Fragment loaders: documented CDNA5 16-bit A-matrix 16x32 layout.
// lanes 0-15 : row M=lane,    K = {0..7, 16..23}  (packed pairs per VGPR)
// lanes 16-31: row M=lane-16, K = {8..15, 24..31}
// Matrix is [16 rows][K] row-major with leading dim ld (halves). B operand
// uses the same layout from a pre-transposed [N][K] matrix.
// ---------------------------------------------------------------------------
__device__ __forceinline__ v16h frag_f16(const _Float16* p, int ld, int k0) {
  int lane = threadIdx.x & 31;
  const _Float16* r = p + (size_t)(lane & 15) * ld + k0 + ((lane >> 4) << 3);
  v16h f;
#pragma unroll
  for (int v = 0; v < 8; ++v) {
    int k = ((v & 4) << 2) + ((v & 3) << 1);   // 0,2,4,6,16,18,20,22
    f[2 * v]     = r[k];
    f[2 * v + 1] = r[k + 1];
  }
  return f;
}

__device__ __forceinline__ v16h frag_f32(const float* p, int ld, int k0) {
  int lane = threadIdx.x & 31;
  const float* r = p + (size_t)(lane & 15) * ld + k0 + ((lane >> 4) << 3);
  v16h f;
#pragma unroll
  for (int v = 0; v < 8; ++v) {
    int k = ((v & 4) << 2) + ((v & 3) << 1);
    f[2 * v]     = (_Float16)r[k];
    f[2 * v + 1] = (_Float16)r[k + 1];
  }
  return f;
}

// C/D 16x16 f32 layout: VGPR e: lanes0-15 -> (M=e, N=lane); lanes16-31 -> (M=8+e).
__device__ __forceinline__ void store_tile(float* C, int ldc, int row0, int col0,
                                           v8f acc, const float* bias, int act) {
  int lane = threadIdx.x & 31;
  int col  = col0 + (lane & 15);
  int rb   = row0 + ((lane >> 4) << 3);
  float bv = bias ? bias[col] : 0.f;
#pragma unroll
  for (int e = 0; e < 8; ++e) {
    float v = acc[e] + bv;
    if (act == 1) v = fmaxf(v, 0.f);
    C[(size_t)(rb + e) * ldc + col] = v;
  }
}

// ---------------------------------------------------------------------------
// Small GEMM (head layers, M == 64): C = act(A_f32[M,K] @ Bt[N,K]^T + bias).
// ---------------------------------------------------------------------------
template <int ACT>
__global__ __launch_bounds__(256) void k_gemm(const float* __restrict__ A,
                                              const _Float16* __restrict__ Bt,
                                              const float* __restrict__ bias,
                                              float* __restrict__ C,
                                              int M, int N, int K) {
  int wv   = threadIdx.x >> 5;
  int trow = wv & 3;
  int tcb  = wv >> 2;
  int row0 = blockIdx.y * 64 + trow * 16;
  int col0 = blockIdx.x * 64 + tcb * 16;
  v8f acc0 = {}, acc1 = {};
  const float*    Ab  = A  + (size_t)row0 * K;
  const _Float16* Bb0 = Bt + (size_t)col0 * K;
  const _Float16* Bb1 = Bt + (size_t)(col0 + 32) * K;
  for (int k0 = 0; k0 < K; k0 += 32) {
    v16h a  = frag_f32(Ab, K, k0);
    acc0 = wmma16(a, frag_f16(Bb0, K, k0), acc0);
    acc1 = wmma16(a, frag_f16(Bb1, K, k0), acc1);
  }
  store_tile(C, N, row0, col0, acc0, bias, ACT);
  store_tile(C, N, row0, col0 + 32, acc1, bias, ACT);
}

// ---------------------------------------------------------------------------
// Main GEMM: 128x128 WG tile, double-buffered LDS staging, f16 A and B.
// 8 waves; wave = (mg = w&3 -> 2 M-tiles, ng = w>>2 -> 4 N-tiles) = 8 accs.
// LDS panels padded to 40 halves/row: row stride 80B -> conflict-free
// ds_load_b128 fragment reads (20*r mod 64 banks all distinct).
// M%128==0, N%128==0, K%32==0.
// ---------------------------------------------------------------------------
#define PAD 40
template <int ACT>
__global__ __launch_bounds__(256) void k_gemm2(const _Float16* __restrict__ A,
                                               const _Float16* __restrict__ Bt,
                                               const float* __restrict__ bias,
                                               float* __restrict__ C,
                                               int M, int N, int K) {
  __shared__ _Float16 Ash[2][128][PAD];
  __shared__ _Float16 Bsh[2][128][PAD];
  const int tid = threadIdx.x, wv = tid >> 5;
  const int mg = wv & 3, ng = wv >> 2;
  const int row0 = blockIdx.y * 128;
  const int col0 = blockIdx.x * 128;

  v8f acc[2][4];
#pragma unroll
  for (int i = 0; i < 2; ++i)
#pragma unroll
    for (int j = 0; j < 4; ++j) acc[i][j] = (v8f){};

  auto stage = [&](int buf, int k0) {
    // 128 rows x 32 halves = 512 x 16B chunks per panel; 2 chunks/thread.
#pragma unroll
    for (int c = tid; c < 512; c += 256) {
      int r = c >> 2, h8 = (c & 3) << 3;
      *reinterpret_cast<uint4*>(&Ash[buf][r][h8]) =
          *reinterpret_cast<const uint4*>(A + (size_t)(row0 + r) * K + k0 + h8);
      *reinterpret_cast<uint4*>(&Bsh[buf][r][h8]) =
          *reinterpret_cast<const uint4*>(Bt + (size_t)(col0 + r) * K + k0 + h8);
    }
  };

  const int nk = K >> 5;
  stage(0, 0);
  __syncthreads();
  for (int kc = 0; kc < nk; ++kc) {
    const int buf = kc & 1;
    if (kc + 1 < nk) stage(buf ^ 1, (kc + 1) << 5);
    v16h af0 = frag_f16(&Ash[buf][mg * 32][0], PAD, 0);
    v16h af1 = frag_f16(&Ash[buf][mg * 32 + 16][0], PAD, 0);
#pragma unroll
    for (int j = 0; j < 4; ++j) {
      v16h bf = frag_f16(&Bsh[buf][ng * 64 + j * 16][0], PAD, 0);
      acc[0][j] = wmma16(af0, bf, acc[0][j]);
      acc[1][j] = wmma16(af1, bf, acc[1][j]);
    }
    __syncthreads();
  }
#pragma unroll
  for (int i = 0; i < 2; ++i)
#pragma unroll
    for (int j = 0; j < 4; ++j)
      store_tile(C, N, row0 + mg * 32 + i * 16, col0 + ng * 64 + j * 16,
                 acc[i][j], bias, ACT);
}

// ---------------------------------------------------------------------------
// Weight cast + transpose: src f32 [K][N] -> dst f16 [N][Kp] (zero pad K..Kp)
// ---------------------------------------------------------------------------
__global__ void k_castT(const float* __restrict__ src, _Float16* __restrict__ dst,
                        int K, int N, int Kp) {
  int i = blockIdx.x * 256 + threadIdx.x;
  if (i >= N * Kp) return;
  int n = i / Kp, k = i - n * Kp;
  dst[i] = (k < K) ? (_Float16)src[(size_t)k * N + n] : (_Float16)0.f;
}

// cgm [M,16] f32 -> [M,32] f16 zero-padded
__global__ void k_padA(const float* __restrict__ src, _Float16* __restrict__ dst,
                       int M) {
  int i = blockIdx.x * 256 + threadIdx.x;
  if (i >= M * 32) return;
  int r = i >> 5, c = i & 31;
  dst[i] = (c < 16) ? (_Float16)src[r * 16 + c] : (_Float16)0.f;
}

__global__ void k_zero(int* p, int n) {
  int i = blockIdx.x * 256 + threadIdx.x;
  if (i < n) p[i] = 0;
}

// ---------------------------------------------------------------------------
// Block reduction (blockDim == 256)
// ---------------------------------------------------------------------------
__device__ __forceinline__ float block_sum(float v) {
  __shared__ float red[256];
  int tid = threadIdx.x;
  __syncthreads();
  red[tid] = v;
  __syncthreads();
#pragma unroll
  for (int off = 128; off > 0; off >>= 1) {
    if (tid < off) red[tid] += red[tid + off];
    __syncthreads();
  }
  return red[0];
}

// LayerNorm over (optionally concatenated) row; optional f32 and f16 outputs.
__global__ __launch_bounds__(256) void k_ln(const float* __restrict__ xa,
                                            const float* __restrict__ xb,
                                            int Ca, int Cb,
                                            const float* __restrict__ s,
                                            const float* __restrict__ bi,
                                            float* __restrict__ out32,
                                            _Float16* __restrict__ out16) {
  int row = blockIdx.x, tid = threadIdx.x;
  int C = Ca + Cb;
  float loc[4];
  int n = 0;
  float sum = 0.f;
  for (int c = tid; c < C; c += 256) {
    float v = (c < Ca) ? xa[(size_t)row * Ca + c] : xb[(size_t)row * Cb + (c - Ca)];
    loc[n++] = v;
    sum += v;
  }
  float mean = block_sum(sum) / C;
  float var = 0.f;
  n = 0;
  for (int c = tid; c < C; c += 256) {
    float d = loc[n++] - mean;
    var += d * d;
  }
  float inv = rsqrtf(block_sum(var) / C + 1e-6f);
  n = 0;
  for (int c = tid; c < C; c += 256) {
    float y = (loc[n++] - mean) * inv * s[c] + bi[c];
    if (out32) out32[(size_t)row * C + c] = y;
    if (out16) out16[(size_t)row * C + c] = (_Float16)y;
  }
}

// x = LN(attn * sigmoid(gpre) + h) over 512
__global__ __launch_bounds__(256) void k_gate_ln(const float* __restrict__ attn,
                                                 const float* __restrict__ gpre,
                                                 const float* __restrict__ hres,
                                                 const float* __restrict__ s,
                                                 const float* __restrict__ bi,
                                                 float* __restrict__ out) {
  int row = blockIdx.x, tid = threadIdx.x;
  float loc[2];
  int n = 0;
  float sum = 0.f;
  for (int c = tid; c < Hh; c += 256) {
    size_t idx = (size_t)row * Hh + c;
    float v = attn[idx] * sigf(gpre[idx]) + hres[idx];
    loc[n++] = v;
    sum += v;
  }
  float mean = block_sum(sum) / Hh;
  float var = 0.f;
  n = 0;
  for (int c = tid; c < Hh; c += 256) {
    float d = loc[n++] - mean;
    var += d * d;
  }
  float inv = rsqrtf(block_sum(var) / Hh + 1e-6f);
  n = 0;
  for (int c = tid; c < Hh; c += 256)
    out[(size_t)row * Hh + c] = (loc[n++] - mean) * inv * s[c] + bi[c];
}

// ---------------------------------------------------------------------------
// LSTM recurrence. grid = (16 col-WGs, 4 batch groups), block 256 (8 waves).
// Wave w computes the 16x16 gate tile (gate g = w>>1, col half = w&1) for h
// columns [32*cw, 32*cw+32). Wht slice lives in 128 VGPRs per wave for the
// whole time loop; h tile staged to LDS as f16 each step. Batch groups are
// independent; the 16 col-WGs of a group sync per step via atomic counters.
// mode 0: gate_fn=tanh, act_fn=sigmoid (bi blocks)  |  mode 1: standard LSTM.
// dir 0: t = s ; dir 1: t = T-1-s (backward pass, equivalent to flip/unflip).
// ---------------------------------------------------------------------------
__global__ __launch_bounds__(256) void k_lstm(const float* __restrict__ xg,
                                              const _Float16* __restrict__ WhT,
                                              float* __restrict__ hout,
                                              int* __restrict__ cnt,
                                              int dir, int mode) {
  __shared__ _Float16 hsh[16][520];
  __shared__ float    csh[16][32];
  __shared__ float    gsh[4][16][32];

  const int bg = blockIdx.y, cw = blockIdx.x;
  const int b0 = bg * 16, hc0 = cw * 32;
  const int tid = threadIdx.x, lane = tid & 31, wv = tid >> 5;
  const int g = wv >> 1, half = wv & 1;
  const int gcol = g * Hh + hc0 + half * 16;

  v16h bf[16];
#pragma unroll
  for (int kc = 0; kc < 16; ++kc)
    bf[kc] = frag_f16(WhT + (size_t)gcol * Hh, Hh, kc * 32);

  for (int i = tid; i < 512; i += 256) csh[i >> 5][i & 31] = 0.f;
  __syncthreads();

  for (int s = 0; s < Tt; ++s) {
    const int t  = dir ? (Tt - 1 - s) : s;
    const int tp = dir ? (t + 1) : (t - 1);

    if (s > 0) {
      for (int i = tid; i < 16 * 512; i += 256) {
        int r = i >> 9, c = i & 511;
        hsh[r][c] = (_Float16)hout[((size_t)(b0 + r) * Tt + tp) * Hh + c];
      }
    }
    __syncthreads();

    v8f acc = {};
    if (s > 0) {
#pragma unroll
      for (int kc = 0; kc < 16; ++kc) {
        v16h a = frag_f16(&hsh[0][0], 520, kc * 32);
        acc = wmma16(a, bf[kc], acc);
      }
    }

    const int tn = dir ? (t - 1) : (t + 1);
    if (tn >= 0 && tn < Tt)
      __builtin_prefetch(&xg[((size_t)(b0 + (lane & 15)) * Tt + tn) * G4 + gcol], 0, 3);

    {
      const int nloc = half * 16 + (lane & 15);
      const int rb   = (lane >> 4) << 3;
#pragma unroll
      for (int e = 0; e < 8; ++e) {
        int m = rb + e;
        float pre = acc[e] +
                    xg[((size_t)(b0 + m) * Tt + t) * G4 + gcol + (lane & 15)];
        float av;
        if (g == 2) av = (mode == 0) ? sigf(pre) : tanhf(pre);   // candidate: act_fn
        else        av = (mode == 0) ? tanhf(pre) : sigf(pre);   // i,f,o: gate_fn
        gsh[g][m][nloc] = av;
      }
    }
    __syncthreads();

    for (int i = tid; i < 512; i += 256) {
      int m = i >> 5, n = i & 31;
      float cn = gsh[1][m][n] * csh[m][n] + gsh[0][m][n] * gsh[2][m][n];
      float ac = (mode == 0) ? sigf(cn) : tanhf(cn);
      csh[m][n] = cn;
      hout[((size_t)(b0 + m) * Tt + t) * Hh + hc0 + n] = gsh[3][m][n] * ac;
    }
    __threadfence();
    __syncthreads();

    if (tid == 0) {
      int* cp = cnt + bg * Tt + s;
      atomicAdd(cp, 1);
      while (__hip_atomic_load(cp, __ATOMIC_ACQUIRE, __HIP_MEMORY_SCOPE_AGENT) < 16)
        __builtin_amdgcn_s_sleep(1);
    }
    __syncthreads();
  }
}

// ---------------------------------------------------------------------------
// Attention: one WG per (b, head, 16-row tile); output written as f16 so the
// wo projection GEMM can consume it directly.
// ---------------------------------------------------------------------------
__global__ __launch_bounds__(64) void k_attn(const float* __restrict__ Q,
                                             const float* __restrict__ Kp,
                                             const float* __restrict__ V,
                                             _Float16* __restrict__ O) {
  __shared__ _Float16 vt[32][304];
  __shared__ float    sc[16][304];
  const int mt = blockIdx.x, h = blockIdx.y, b = blockIdx.z;
  const int tid = threadIdx.x, lane = tid & 31, wv = tid >> 5;
  const size_t base = (size_t)b * Tt * 128 + h * HDd;

  for (int i = tid; i < Tt * HDd; i += 64) {
    int t = i >> 5, d = i & 31;
    vt[d][t] = (_Float16)V[base + (size_t)t * 128 + d];
  }
  __syncthreads();

  v16h aq = frag_f32(Q + base + (size_t)(mt * 16) * 128, 128, 0);
  const float scale = 0.17677669529663687f;   // 1/sqrt(32)
  for (int nt = wv * 9; nt < wv * 9 + 9; ++nt) {
    v16h bk = frag_f32(Kp + base + (size_t)(nt * 16) * 128, 128, 0);
    v8f acc = {};
    acc = wmma16(aq, bk, acc);
    int col = nt * 16 + (lane & 15);
    int rb  = (lane >> 4) << 3;
#pragma unroll
    for (int e = 0; e < 8; ++e) sc[rb + e][col] = acc[e] * scale;
  }
  __syncthreads();

  if (tid < 16) {
    float mx = -1e30f;
    for (int j = 0; j < Tt; ++j) mx = fmaxf(mx, sc[tid][j]);
    float sum = 0.f;
    for (int j = 0; j < Tt; ++j) {
      float e = __expf(sc[tid][j] - mx);
      sc[tid][j] = e;
      sum += e;
    }
    float inv = 1.f / sum;
    for (int j = 0; j < Tt; ++j) sc[tid][j] *= inv;
  }
  __syncthreads();

  v8f acc = {};
  for (int kc = 0; kc < 9; ++kc) {
    v16h a = frag_f32(&sc[0][0], 304, kc * 32);
    v16h b = frag_f16(&vt[wv * 16][0], 304, kc * 32);
    acc = wmma16(a, b, acc);
  }
  int d  = wv * 16 + (lane & 15);
  int rb = (lane >> 4) << 3;
#pragma unroll
  for (int e = 0; e < 8; ++e)
    O[base + (size_t)(mt * 16 + rb + e) * 128 + d] = (_Float16)acc[e];
}

// mean/max pool over T + concat other -> Y [B, 1088]
__global__ __launch_bounds__(512) void k_pool(const float* __restrict__ X,
                                              const float* __restrict__ other,
                                              float* __restrict__ Y) {
  int b = blockIdx.x, c = threadIdx.x;
  float sum = 0.f, mx = -1e30f;
  for (int t = 0; t < Tt; ++t) {
    float v = X[((size_t)b * Tt + t) * Hh + c];
    sum += v;
    mx = fmaxf(mx, v);
  }
  Y[(size_t)b * 1088 + c]      = sum * (1.f / Tt);
  Y[(size_t)b * 1088 + Hh + c] = mx;
  if (c < FOTH) Y[(size_t)b * 1088 + 1024 + c] = other[b * FOTH + c];
}

__global__ void k_final(const float* __restrict__ h2, const float* __restrict__ w,
                        const float* __restrict__ b, float* __restrict__ out) {
  int i = threadIdx.x;
  if (i >= Bb) return;
  float s = 0.f;
  for (int j = 0; j < D2d; ++j) s += h2[i * D2d + j] * w[j];
  out[i] = s + b[0];
}

// ---------------------------------------------------------------------------
// Host orchestration
// ---------------------------------------------------------------------------
static inline char* bump(char*& p, size_t bytes) {
  char* r = p;
  p += (bytes + 255) & ~(size_t)255;
  return r;
}

extern "C" void kernel_launch(void* const* d_in, const int* in_sizes, int n_in,
                              void* d_out, int out_size, void* d_ws, size_t ws_size,
                              hipStream_t stream) {
  (void)in_sizes; (void)n_in; (void)out_size; (void)ws_size;
  auto F = [&](int i) { return (const float*)d_in[i]; };
  const float* cgm = F(0);
  const float* oth = F(1);

  char* wp = (char*)d_ws;
  // f16 transposed weights
  _Float16* d0T = (_Float16*)bump(wp, (size_t)Hh * 32 * 2);
  _Float16* wxT[5];
  _Float16* whT[5];
  const int wx_in[5] = {Hh, Hh, 2 * Hh, 2 * Hh, 2 * Hh};
  for (int i = 0; i < 5; ++i) {
    wxT[i] = (_Float16*)bump(wp, (size_t)G4 * wx_in[i] * 2);
    whT[i] = (_Float16*)bump(wp, (size_t)G4 * Hh * 2);
  }
  _Float16* wqT = (_Float16*)bump(wp, (size_t)128 * Hh * 2);
  _Float16* wkT = (_Float16*)bump(wp, (size_t)128 * Hh * 2);
  _Float16* wvT = (_Float16*)bump(wp, (size_t)128 * Hh * 2);
  _Float16* woT = (_Float16*)bump(wp, (size_t)Hh * 128 * 2);
  _Float16* gwT = (_Float16*)bump(wp, (size_t)Hh * Hh * 2);
  _Float16* d1T = (_Float16*)bump(wp, (size_t)D1d * 1088 * 2);
  _Float16* d2T = (_Float16*)bump(wp, (size_t)D2d * D1d * 2);
  // activations
  _Float16* Apad = (_Float16*)bump(wp, (size_t)ROWS * 32 * 2);
  float*    X0   = (float*)bump(wp, (size_t)ROWS * Hh * 4);      // pre-LN d0 out
  _Float16* Xh   = (_Float16*)bump(wp, (size_t)ROWS * Hh * 2);   // LN(d0) f16
  float*    XG   = (float*)bump(wp, (size_t)ROWS * G4 * 4);      // xg / gate_pre
  float*    FO   = (float*)bump(wp, (size_t)ROWS * Hh * 4);
  float*    BO   = (float*)bump(wp, (size_t)ROWS * Hh * 4);
  _Float16* X1h  = (_Float16*)bump(wp, (size_t)ROWS * 2 * Hh * 2);
  _Float16* X2h  = (_Float16*)bump(wp, (size_t)ROWS * 2 * Hh * 2);
  float*    HL   = (float*)bump(wp, (size_t)ROWS * Hh * 4);
  float*    HLN  = (float*)bump(wp, (size_t)ROWS * Hh * 4);
  _Float16* HLNh = (_Float16*)bump(wp, (size_t)ROWS * Hh * 2);
  float*    Qb   = (float*)bump(wp, (size_t)ROWS * 128 * 4);
  float*    Kb   = (float*)bump(wp, (size_t)ROWS * 128 * 4);
  float*    Vb   = (float*)bump(wp, (size_t)ROWS * 128 * 4);
  _Float16* Obh  = (_Float16*)bump(wp, (size_t)ROWS * 128 * 2);
  float*    ATT  = (float*)bump(wp, (size_t)ROWS * Hh * 4);
  float*    XF   = (float*)bump(wp, (size_t)ROWS * Hh * 4);
  float*    Y    = (float*)bump(wp, (size_t)Bb * 1088 * 4);
  float*    H1   = (float*)bump(wp, (size_t)Bb * D1d * 4);
  float*    H2   = (float*)bump(wp, (size_t)Bb * D2d * 4);
  int*      CNT  = (int*)bump(wp, (size_t)5 * 4 * Tt * 4);

  const dim3 blk(256);
  auto castT = [&](const float* src, _Float16* dst, int K, int N, int Kp) {
    int n = N * Kp;
    k_castT<<<dim3((n + 255) / 256), blk, 0, stream>>>(src, dst, K, N, Kp);
  };
  // big GEMM: f16 A, 128x128 tiles
  auto gemm2 = [&](const _Float16* A, const _Float16* Bt, const float* bias,
                   float* C, int M, int N, int K) {
    k_gemm2<0><<<dim3(N / 128, M / 128), blk, 0, stream>>>(A, Bt, bias, C, M, N, K);
  };
  // small head GEMM: f32 A, 64x64 tiles
  auto gemm = [&](const float* A, const _Float16* Bt, const float* bias, float* C,
                  int M, int N, int K, int act) {
    dim3 grid(N / 64, M / 64);
    if (act) k_gemm<1><<<grid, blk, 0, stream>>>(A, Bt, bias, C, M, N, K);
    else     k_gemm<0><<<grid, blk, 0, stream>>>(A, Bt, bias, C, M, N, K);
  };

  // -- weight prep ---------------------------------------------------------
  k_zero<<<dim3((5 * 4 * Tt + 255) / 256), blk, 0, stream>>>(CNT, 5 * 4 * Tt);
  castT(F(2), d0T, 16, Hh, 32);
  const int wx_idx[5] = {6, 9, 14, 17, 22};
  const int wh_idx[5] = {7, 10, 15, 18, 23};
  for (int i = 0; i < 5; ++i) {
    castT(F(wx_idx[i]), wxT[i], wx_in[i], G4, wx_in[i]);
    castT(F(wh_idx[i]), whT[i], Hh, G4, Hh);
  }
  castT(F(27), wqT, Hh, 128, Hh);
  castT(F(29), wkT, Hh, 128, Hh);
  castT(F(31), wvT, Hh, 128, Hh);
  castT(F(33), woT, 128, Hh, 128);
  castT(F(35), gwT, Hh, Hh, Hh);
  castT(F(39), d1T, 1088, D1d, 1088);
  castT(F(43), d2T, D1d, D2d, D1d);

  // -- stem: x = LN(d0(cgm)) ----------------------------------------------
  k_padA<<<dim3((ROWS * 32 + 255) / 256), blk, 0, stream>>>(cgm, Apad, ROWS);
  gemm2(Apad, d0T, F(3), X0, ROWS, Hh, 32);
  k_ln<<<dim3(ROWS), blk, 0, stream>>>(X0, nullptr, Hh, 0, F(4), F(5), nullptr, Xh);

  const dim3 lgrid(16, 4);
  // -- bi0 -----------------------------------------------------------------
  gemm2(Xh, wxT[0], F(8), XG, ROWS, G4, Hh);
  k_lstm<<<lgrid, blk, 0, stream>>>(XG, whT[0], FO, CNT + 0 * 4 * Tt, 0, 0);
  gemm2(Xh, wxT[1], F(11), XG, ROWS, G4, Hh);
  k_lstm<<<lgrid, blk, 0, stream>>>(XG, whT[1], BO, CNT + 1 * 4 * Tt, 1, 0);
  k_ln<<<dim3(ROWS), blk, 0, stream>>>(FO, BO, Hh, Hh, F(12), F(13), nullptr, X1h);
  // -- bi1 -----------------------------------------------------------------
  gemm2(X1h, wxT[2], F(16), XG, ROWS, G4, 2 * Hh);
  k_lstm<<<lgrid, blk, 0, stream>>>(XG, whT[2], FO, CNT + 2 * 4 * Tt, 0, 0);
  gemm2(X1h, wxT[3], F(19), XG, ROWS, G4, 2 * Hh);
  k_lstm<<<lgrid, blk, 0, stream>>>(XG, whT[3], BO, CNT + 3 * 4 * Tt, 1, 0);
  k_ln<<<dim3(ROWS), blk, 0, stream>>>(FO, BO, Hh, Hh, F(20), F(21), nullptr, X2h);
  // -- attention block -----------------------------------------------------
  gemm2(X2h, wxT[4], F(24), XG, ROWS, G4, 2 * Hh);
  k_lstm<<<lgrid, blk, 0, stream>>>(XG, whT[4], HL, CNT + 4 * 4 * Tt, 0, 1);
  k_ln<<<dim3(ROWS), blk, 0, stream>>>(HL, nullptr, Hh, 0, F(25), F(26), HLN, HLNh);
  gemm2(HLNh, wqT, F(28), Qb, ROWS, 128, Hh);
  gemm2(HLNh, wkT, F(30), Kb, ROWS, 128, Hh);
  gemm2(HLNh, wvT, F(32), Vb, ROWS, 128, Hh);
  k_attn<<<dim3(Tt / 16, NHd, Bb), dim3(64), 0, stream>>>(Qb, Kb, Vb, Obh);
  gemm2(Obh, woT, F(34), ATT, ROWS, Hh, 128);
  gemm2(HLNh, gwT, F(36), XG, ROWS, Hh, Hh);          // gate pre-act (reuse XG)
  k_gate_ln<<<dim3(ROWS), blk, 0, stream>>>(ATT, XG, HLN, F(37), F(38), XF);
  // -- head ----------------------------------------------------------------
  k_pool<<<dim3(Bb), dim3(512), 0, stream>>>(XF, oth, Y);
  gemm(Y, d1T, F(40), H1, Bb, D1d, 1088, 1);
  k_ln<<<dim3(Bb), blk, 0, stream>>>(H1, nullptr, D1d, 0, F(41), F(42), H1, nullptr);
  gemm(H1, d2T, F(44), H2, Bb, D2d, D1d, 1);
  k_ln<<<dim3(Bb), blk, 0, stream>>>(H2, nullptr, D2d, 0, F(45), F(46), H2, nullptr);
  k_final<<<dim3(1), dim3(64), 0, stream>>>(H2, F(47), F(48), (float*)d_out);
}
